// SlefAttention_46213848105807
// MI455X (gfx1250) — matrix operand order repaired and checked by reading, not compiled, not run on hardware
//
#include <hip/hip_runtime.h>
#include <hip/hip_bf16.h>

// CDNA5 (gfx1250, wave32) flash-attention forward with masking + dropout.
// Transpose-free formulation: S^T = K @ Q^T (K row-major A-operand, Q^T
// B-fragments built once), then O = P @ V (P scratch written with b128
// stores since S^T's C-layout is key-contiguous per lane). bf16 WMMA in
// the log2 domain, f32 accumulation, wave-shared running max with a
// wave-uniform skip of the rescale, and 8 independent denominator
// partial accumulators to avoid serial VALU dependency chains.

typedef __attribute__((ext_vector_type(16))) __bf16 v16bf;
typedef __attribute__((ext_vector_type(8)))  float  v8f;

union AFrag { v16bf v; uint4 q[2]; __bf16 e[16]; };
union CFrag { v8f   v; float f[8]; };
union Pack8 { __bf16 e[8]; uint4 u; };

#define SS    2048
#define DH    128
#define NT    64      // key tile
#define WAVES 8

__global__ __launch_bounds__(256, 1)
void attn_fwd_kernel(const float* __restrict__ Q,
                     const float* __restrict__ K,
                     const float* __restrict__ V,
                     const long long* __restrict__ valid_lens,
                     float* __restrict__ Out) {
    // 48KB: Ks[64][128] | Vs[64][128] | Ps[8 waves][16][64]
    // prologue reuses the Ks+Vs region (32KB) to stage Q (128x128 bf16)
    __shared__ __align__(16) __bf16 smem[24576];
    const int tid  = threadIdx.x;
    const int wave = tid >> 5;
    const int lane = tid & 31;
    const int m    = lane & 15;          // A row / C column index
    const int h    = lane >> 4;          // lane half

    __bf16 (*Ks)[DH] = (__bf16(*)[DH])smem;                    // [64][128]
    __bf16 (*Vs)[DH] = (__bf16(*)[DH])(smem + NT * DH);        // [64][128]
    __bf16 (*Ps)[NT] = (__bf16(*)[NT])(smem + 2 * NT * DH + wave * 16 * NT);

    const int b     = blockIdx.x >> 4;           // 16 q-tiles per batch
    const int qtile = blockIdx.x & 15;
    const int q0    = qtile * 128 + wave * 16;   // this wave's 16 query rows

    const int vl  = (int)valid_lens[b];
    const int nkt = (vl + NT - 1) / NT;          // key tiles to process

    const float sc = 0.0625f * 1.44269504088896f;   // 1/sqrt(2*128) * log2(e)

    // ---- prologue: stage all 128 block queries to LDS as scaled bf16 ----
#pragma unroll
    for (int i = 0; i < 8; ++i) {
        const int c = tid + i * 256;     // 2048 chunks of 8 floats
        const int q = c >> 4;
        const int d = (c & 15) * 8;
        const float* qp = Q + ((size_t)(b * SS + qtile * 128 + q)) * DH + d;
        float4 a0 = *(const float4*)qp;
        float4 a1 = *(const float4*)(qp + 4);
        Pack8 pk;
        pk.e[0] = (__bf16)(a0.x * sc); pk.e[1] = (__bf16)(a0.y * sc);
        pk.e[2] = (__bf16)(a0.z * sc); pk.e[3] = (__bf16)(a0.w * sc);
        pk.e[4] = (__bf16)(a1.x * sc); pk.e[5] = (__bf16)(a1.y * sc);
        pk.e[6] = (__bf16)(a1.z * sc); pk.e[7] = (__bf16)(a1.w * sc);
        *(uint4*)&smem[q * DH + d] = pk.u;
    }
    __syncthreads();

    // ---- build Q^T B-fragments (32 d-rows x 16 queries each), once ----
    AFrag qb[4];
#pragma unroll
    for (int s = 0; s < 4; ++s)
#pragma unroll
        for (int q = 0; q < 16; ++q)
            qb[s].e[q] = smem[(wave * 16 + m) * DH + s * 32 + 8 * h + q + (q & 8)];
    __syncthreads();

    // ---- state ----
    float Mrun = -3.0e38f;
    float lpv[8];                        // independent partial denominators
    CFrag acc[8];
#pragma unroll
    for (int v = 0; v < 8; ++v) {
        lpv[v] = 0.0f;
#pragma unroll
        for (int j = 0; j < 8; ++j) acc[v].f[j] = 0.0f;
    }

    for (int t = 0; t < nkt; ++t) {
        const int kbase = t * NT;

        // ---- cooperative load: K and V tiles, both row-major bf16 ----
#pragma unroll
        for (int i = 0; i < 4; ++i) {
            const int c   = tid + i * 256;   // 1024 chunks of 8 floats
            const int key = c >> 4;
            const int d   = (c & 15) * 8;
            const size_t goff = ((size_t)(b * SS + kbase + key)) * DH + d;
            float4 k0 = *(const float4*)(K + goff);
            float4 k1 = *(const float4*)(K + goff + 4);
            Pack8 pk;
            pk.e[0] = (__bf16)k0.x; pk.e[1] = (__bf16)k0.y;
            pk.e[2] = (__bf16)k0.z; pk.e[3] = (__bf16)k0.w;
            pk.e[4] = (__bf16)k1.x; pk.e[5] = (__bf16)k1.y;
            pk.e[6] = (__bf16)k1.z; pk.e[7] = (__bf16)k1.w;
            *(uint4*)&Ks[key][d] = pk.u;
            float4 v0 = *(const float4*)(V + goff);
            float4 v1 = *(const float4*)(V + goff + 4);
            Pack8 pv;
            pv.e[0] = (__bf16)v0.x; pv.e[1] = (__bf16)v0.y;
            pv.e[2] = (__bf16)v0.z; pv.e[3] = (__bf16)v0.w;
            pv.e[4] = (__bf16)v1.x; pv.e[5] = (__bf16)v1.y;
            pv.e[6] = (__bf16)v1.z; pv.e[7] = (__bf16)v1.w;
            *(uint4*)&Vs[key][d] = pv.u;
        }
        // prefetch next tile: 256 threads x 128B lines == one 32KB tile each
        if (t + 1 < nkt) {
            const size_t nb = ((size_t)(b * SS + kbase + NT)) * DH + tid * 32;
            __builtin_prefetch(K + nb, 0, 1);
            __builtin_prefetch(V + nb, 0, 1);
        }
        __syncthreads();

        // ---- S^T = K @ Q^T : rows=keys (4 chunks of 16), cols=queries ----
        CFrag sfrag[4];
#pragma unroll
        for (int c4 = 0; c4 < 4; ++c4) {
            CFrag s;
#pragma unroll
            for (int j = 0; j < 8; ++j) s.f[j] = 0.0f;
#pragma unroll
            for (int s4 = 0; s4 < 4; ++s4) {
                AFrag aa;
                aa.q[0] = *(const uint4*)&Ks[c4 * 16 + m][s4 * 32 + 8 * h];
                aa.q[1] = *(const uint4*)&Ks[c4 * 16 + m][s4 * 32 + 16 + 8 * h];
                s.v = __builtin_amdgcn_wmma_f32_16x16x32_bf16(
                          false, aa.v, false, qb[s4].v, (short)0, s.v, false, false);
            }
            sfrag[c4] = s;
        }

        // ---- mask (boundary tile only) + wave-shared max ----
        float tm = -3.0e38f;
        if (kbase + NT <= vl) {
#pragma unroll
            for (int c4 = 0; c4 < 4; ++c4)
#pragma unroll
                for (int v = 0; v < 8; ++v)
                    tm = fmaxf(tm, sfrag[c4].f[v]);
        } else {
            const int lim = vl - kbase - 8 * h;   // key-in-tile bound, per half
#pragma unroll
            for (int c4 = 0; c4 < 4; ++c4)
#pragma unroll
                for (int v = 0; v < 8; ++v) {
                    float x = (c4 * 16 + v < lim) ? sfrag[c4].f[v] : -1.0e6f;
                    sfrag[c4].f[v] = x;
                    tm = fmaxf(tm, x);
                }
        }
#pragma unroll
        for (int off = 1; off < 32; off <<= 1)
            tm = fmaxf(tm, __shfl_xor(tm, off, 32));

        // ---- rescale only when the wave-uniform max grows (rare) ----
        if (tm > Mrun) {                  // uniform across wave -> scalar branch
            const float f = __builtin_amdgcn_exp2f(Mrun - tm);
            Mrun = tm;
#pragma unroll
            for (int v = 0; v < 8; ++v) lpv[v] *= f;
#pragma unroll
            for (int d8 = 0; d8 < 8; ++d8)
#pragma unroll
                for (int v = 0; v < 8; ++v) acc[d8].f[v] *= f;
        }

        // ---- P = exp2(S - M); dropout; vectorized b128 stores to P scratch ----
        // lane = query m; keys along v are contiguous in Ps[m][...]
        const unsigned ib = ((unsigned)(b * SS + q0 + m)) * (unsigned)SS
                          + (unsigned)(kbase + 8 * h);
        const unsigned xb = ib * 0x9E3779B1u + 0xB5297A4Du;
#pragma unroll
        for (int c4 = 0; c4 < 4; ++c4) {
            Pack8 pw;
#pragma unroll
            for (int v = 0; v < 8; ++v) {
                const float p = __builtin_amdgcn_exp2f(sfrag[c4].f[v] - Mrun);
                lpv[v] += p;              // 8 independent chains, depth 4/tile
                const unsigned xv = xb + (unsigned)(c4 * 16 + v) * 0x9E3779B1u;
                const unsigned y  = xv ^ (xv >> 15);
                const float pd = (y < 3865470566u) ? p * 1.11111111f : 0.0f;
                pw.e[v] = (__bf16)pd;
            }
            *(uint4*)&Ps[m][c4 * 16 + 8 * h] = pw.u;
        }

        // ---- O += P_drop @ V ----
        AFrag pa[2];
#pragma unroll
        for (int s2 = 0; s2 < 2; ++s2) {
            const int k0 = s2 * 32;
            pa[s2].q[0] = *(const uint4*)&Ps[m][k0 + 8 * h];
            pa[s2].q[1] = *(const uint4*)&Ps[m][k0 + 16 + 8 * h];
        }
#pragma unroll
        for (int d8 = 0; d8 < 8; ++d8) {
#pragma unroll
            for (int s2 = 0; s2 < 2; ++s2) {
                AFrag bv;
#pragma unroll
                for (int e = 0; e < 16; ++e)
                    bv.e[e] = Vs[s2 * 32 + 8 * h + e + (e & 8)][d8 * 16 + m];
                acc[d8].v = __builtin_amdgcn_wmma_f32_16x16x32_bf16(
                                false, pa[s2].v, false, bv.v, (short)0, acc[d8].v,
                                false, false);
            }
        }
        __syncthreads();   // protect Ks/Vs before next tile's loads
    }

    // ---- final: tree-fold partials, reduce, redistribute per-row, store ----
    float lp = ((lpv[0] + lpv[1]) + (lpv[2] + lpv[3]))
             + ((lpv[4] + lpv[5]) + (lpv[6] + lpv[7]));
    lp += __shfl_xor(lp, 16, 32);        // both halves of each query column
    float* lred = (float*)Ps;            // reuse per-wave scratch (64B)
    if (h == 0) lred[m] = 1.0f / lp;     // inverse denom for query m
    __syncthreads();

    float* obase = Out + ((size_t)(b * SS + q0)) * DH;
#pragma unroll
    for (int v = 0; v < 8; ++v) {
        const float inv = lred[v + 8 * h];
        const int r = v + 8 * h;
#pragma unroll
        for (int d8 = 0; d8 < 8; ++d8)
            obase[(size_t)r * DH + d8 * 16 + m] = acc[d8].f[v] * inv;
    }
}

extern "C" void kernel_launch(void* const* d_in, const int* in_sizes, int n_in,
                              void* d_out, int out_size, void* d_ws, size_t ws_size,
                              hipStream_t stream) {
    (void)in_sizes; (void)n_in; (void)out_size; (void)d_ws; (void)ws_size;
    const float* Q = (const float*)d_in[0];
    const float* K = (const float*)d_in[1];
    const float* V = (const float*)d_in[2];
    const long long* vl = (const long long*)d_in[3];
    float* out = (float*)d_out;
    dim3 grid(16 * 16);   // B * (S / 128)
    dim3 block(256);      // 8 wave32 waves
    hipLaunchKernelGGL(attn_fwd_kernel, grid, block, 0, stream, Q, K, V, vl, out);
}
